// MyModel_69715909148737
// MI455X (gfx1250) — matrix operand order repaired and checked by reading, not compiled, not run on hardware
//
#include <hip/hip_runtime.h>
#include <hip/hip_fp16.h>
#include <stdint.h>

typedef _Float16 v16h __attribute__((ext_vector_type(16)));
typedef _Float16 v8h  __attribute__((ext_vector_type(8)));
typedef float    v8f  __attribute__((ext_vector_type(8)));
typedef unsigned int u32x4 __attribute__((ext_vector_type(4)));
typedef int          i32x4 __attribute__((ext_vector_type(4)));
typedef int          i32x8 __attribute__((ext_vector_type(8)));

#define VOCAB 10000
#define EMB   256
#define UNITS 128
#define GATES 512   /* 4*UNITS */
#define BATCH 256
#define TLEN  1024

// Tensor Data Mover path only when the builtin exists on this pass.
#if defined(__gfx1250__) && __has_builtin(__builtin_amdgcn_tensor_load_to_lds)
#define USE_TDM 1
#else
#define USE_TDM 0
#endif

// Concatenate two contiguous 8x f16 chunks into one WMMA operand.
__device__ __forceinline__ v16h load_frag16(const _Float16* p0, const _Float16* p1) {
    v8h lo = *(const v8h*)p0;
    v8h hi = *(const v8h*)p1;
    return __builtin_shufflevector(lo, hi, 0,1,2,3,4,5,6,7,8,9,10,11,12,13,14,15);
}

__device__ __forceinline__ float fast_sigmoid(float z) {
    return __builtin_amdgcn_rcpf(1.f + __expf(-z));   // v_exp + v_rcp
}

#if USE_TDM
// 1-D contiguous f16 run -> LDS (D# per CDNA5 ISA).
__device__ __forceinline__ void tdm_load_f16_1d(uint32_t lds_byte_off,
                                                const void* gaddr,
                                                uint32_t nelem) {
    uint64_t ga = (uint64_t)(uintptr_t)gaddr;
    u32x4 g0;
    g0[0] = 1u;                                               // count=1
    g0[1] = lds_byte_off;                                     // lds_addr
    g0[2] = (uint32_t)(ga & 0xffffffffu);                     // addr[31:0]
    g0[3] = (uint32_t)((ga >> 32) & 0x1ffffffu) | (2u << 30); // addr[56:32]|type=2
    i32x8 g1;
    g1[0] = (int)(1u << 16);                    // data_size=1 (2B)
    g1[1] = (int)((nelem & 0xffffu) << 16);     // tensor_dim0[15:0]
    g1[2] = (int)((nelem >> 16) & 0xffffu);     // tensor_dim0[31:16]
    g1[3] = (int)((nelem & 0xffffu) << 16);     // tile_dim0
    g1[4] = 0;                                  // tile_dim1/2 unused
    g1[5] = (int)nelem;                         // dim0 stride
    g1[6] = 0;
    g1[7] = 0;
    i32x4 z4 = {0, 0, 0, 0};
#if defined(__clang_major__) && __clang_major__ >= 23
    i32x8 z8 = {0, 0, 0, 0, 0, 0, 0, 0};
    __builtin_amdgcn_tensor_load_to_lds(g0, g1, z4, z4, z8, 0);
#else
    __builtin_amdgcn_tensor_load_to_lds(g0, g1, z4, z4, 0);
#endif
}

// TDM gather: pull 16 rows (16-bit indices in groups 2/3) of a 2-D f16
// tensor into LDS, compacted. Row width = row_elems, nrows for OOB bound.
__device__ __forceinline__ void tdm_gather16_f16(uint32_t lds_byte_off,
                                                 const void* gbase,
                                                 uint32_t row_elems,
                                                 uint32_t nrows,
                                                 i32x4 idxlo, i32x4 idxhi) {
    uint64_t ga = (uint64_t)(uintptr_t)gbase;
    u32x4 g0;
    g0[0] = 1u | (1u << 31);                                  // count=1|gather, 16b idx
    g0[1] = lds_byte_off;
    g0[2] = (uint32_t)(ga & 0xffffffffu);
    g0[3] = (uint32_t)((ga >> 32) & 0x1ffffffu) | (2u << 30);
    i32x8 g1;
    g1[0] = (int)(1u << 16);                                  // data_size=1 (2B)
    g1[1] = (int)((row_elems & 0xffffu) << 16);               // tensor_dim0 lo
    g1[2] = (int)(((row_elems >> 16) & 0xffffu) |
                  ((nrows & 0xffffu) << 16));                 // dim0 hi | dim1 lo
    g1[3] = (int)(((nrows >> 16) & 0xffffu) |
                  ((row_elems & 0xffffu) << 16));             // dim1 hi | tile_dim0
    g1[4] = 16;                                               // tile_dim1 = #indices
    g1[5] = (int)row_elems;                                   // dim0 stride
    g1[6] = 0;
    g1[7] = 0;
#if defined(__clang_major__) && __clang_major__ >= 23
    i32x8 z8 = {0, 0, 0, 0, 0, 0, 0, 0};
    __builtin_amdgcn_tensor_load_to_lds(g0, g1, idxlo, idxhi, z8, 0);
#else
    __builtin_amdgcn_tensor_load_to_lds(g0, g1, idxlo, idxhi, 0);
#endif
}
#endif

// ---------------------------------------------------------------------------
// Kernel 0: f32 -> f16 conversion (Wk once, emb table once) so all kernel-1
// staging is raw f16 data the Tensor Data Mover can DMA.
// ---------------------------------------------------------------------------
__global__ void __launch_bounds__(256)
f32_to_f16_kernel(const float* __restrict__ src, _Float16* __restrict__ dst,
                  int n) {
    int i = blockIdx.x * 256 + threadIdx.x;
    if (i < n) dst[i] = (_Float16)src[i];
}

// ---------------------------------------------------------------------------
// Kernel 1: xg[T][B][512] = f16( gather(emb, x) @ Wk + bias )
// Block: 8 wave32, tile = 64 rows (one b, 64 consecutive t) x 512 cols.
// A: TDM gather of 16 embedding rows per descriptor (waves 0-3).
// B: TDM 32KB tile per k-step (wave 0). Fallbacks: plain f16 copy loops.
// Epilogue: transpose through LDS -> coalesced global_store_b128.
// ---------------------------------------------------------------------------
__global__ void __launch_bounds__(256)
lstm_xgemm_kernel(const int* __restrict__ x,
                  const _Float16* __restrict__ embh,
                  const _Float16* __restrict__ Wkh,
                  const float* __restrict__ bias,
                  _Float16* __restrict__ xg)
{
    extern __shared__ _Float16 smem[];
    _Float16* As = smem;              // [64][EMB]   = 32KB
    _Float16* Bs = smem + 64 * EMB;   // [32][GATES] = 32KB

    const int  tid   = threadIdx.x;
    const int  lane  = tid & 31;
    const int  w     = tid >> 5;
    const long mbase = (long)blockIdx.x * 64;     // M = b*T + t
    const int  bidx  = (int)(mbase / TLEN);       // 64 | 1024 -> single b/block
    const int  t0    = (int)(mbase % TLEN);

    // ---- Stage A: 64 embedding rows (f16) ----
#if USE_TDM
    if (w < 4) {   // 4 gather descriptors x 16 rows each
        const int rb = (int)mbase + w * 16;
        i32x4 ilo, ihi;
        #pragma unroll
        for (int j = 0; j < 4; ++j) {
            ilo[j] = (x[rb + 2 * j] & 0xffff)     | (x[rb + 2 * j + 1] << 16);
            ihi[j] = (x[rb + 8 + 2 * j] & 0xffff) | (x[rb + 8 + 2 * j + 1] << 16);
        }
        uint32_t lds_off = (uint32_t)(uintptr_t)(void*)(As + w * 16 * EMB);
        tdm_gather16_f16(lds_off, embh, EMB, VOCAB, ilo, ihi);
        __builtin_amdgcn_s_wait_tensorcnt(0);
    }
#else
    for (int i = tid; i < 64 * EMB; i += 256) {
        int r = i >> 8;
        int k = i & 255;
        As[i] = embh[(long)x[mbase + r] * EMB + k];
    }
#endif

    const int m  = lane & 15;
    const int hk = (lane >> 4) * 8;
    const int n  = lane & 15;

    v8f acc[4][4];
    #pragma unroll
    for (int nt = 0; nt < 4; ++nt) {
        float bv = bias[w * 64 + nt * 16 + n];
        #pragma unroll
        for (int mt = 0; mt < 4; ++mt)
            #pragma unroll
            for (int r = 0; r < 8; ++r) acc[mt][nt][r] = bv;
    }

    for (int kt = 0; kt < 8; ++kt) {
        __syncthreads();   // prior Bs reads complete (covers A staging on kt=0)
#if USE_TDM
        if (w == 0) {      // one wave launches the DMA
            uint32_t lds_off = (uint32_t)(uintptr_t)(void*)Bs;
            tdm_load_f16_1d(lds_off, Wkh + kt * 32 * GATES, 32 * GATES);
            __builtin_amdgcn_s_wait_tensorcnt(0);
        }
#else
        for (int i = tid; i < 32 * GATES; i += 256)
            Bs[i] = Wkh[kt * 32 * GATES + i];
#endif
        __syncthreads();

        v16h bfrag[4];
        #pragma unroll
        for (int nt = 0; nt < 4; ++nt) {   // B: lane%16 = N col, elements = K (table order)
            #pragma unroll
            for (int s = 0; s < 16; ++s) {
                int kk = (s < 8) ? (hk + s) : (16 + hk + (s - 8));
                bfrag[nt][s] = Bs[kk * GATES + w * 64 + nt * 16 + n];
            }
        }
        #pragma unroll
        for (int mt = 0; mt < 4; ++mt) {
            const _Float16* ap = &As[(mt * 16 + m) * EMB + kt * 32 + hk];
            v16h afrag = load_frag16(ap, ap + 16);
            #pragma unroll
            for (int nt = 0; nt < 4; ++nt)
                acc[mt][nt] = __builtin_amdgcn_wmma_f32_16x16x32_f16(
                    false, afrag, false, bfrag[nt], (short)0, acc[mt][nt],
                    false, false);
        }
    }

    // ---- Epilogue: LDS transpose, then coalesced b128 stores ----
    __syncthreads();                       // staging buffers dead; reuse 64KB
    _Float16* Os = smem;                   // [64][GATES]
    const int rowoff = (lane < 16) ? 0 : 8;
    #pragma unroll
    for (int mt = 0; mt < 4; ++mt)
        #pragma unroll
        for (int nt = 0; nt < 4; ++nt)
            #pragma unroll
            for (int r = 0; r < 8; ++r)
                Os[(mt * 16 + r + rowoff) * GATES + (w * 64 + nt * 16 + n)] =
                    (_Float16)acc[mt][nt][r];
    __syncthreads();

    // 64 rows x 512 cols -> [T][B][512]; 16 x global_store_b128 per thread.
    for (int i = tid; i < (64 * GATES) / 8; i += 256) {
        int lr  = (i * 8) >> 9;            // local row (t offset)
        int col = (i * 8) & 511;
        int t   = t0 + lr;
        *(v8h*)&xg[((long)t * BATCH + bidx) * GATES + col] =
            *(const v8h*)&Os[i * 8];
    }
}

// ---------------------------------------------------------------------------
// Kernel 2: sequential recurrence. 16 blocks x 16 batch rows.
// Wave w owns hidden units [w*16, w*16+16) and computes ALL FOUR gate tiles
// for them -> gates are register-local, no z exchange. h double-buffered in
// LDS -> exactly one barrier per step. Wr frags + c state live in registers.
// ---------------------------------------------------------------------------
#define HPAD (UNITS + 8)

__global__ void __launch_bounds__(256)
lstm_recur_kernel(const _Float16* __restrict__ xg,   // [T][B][512]
                  const float* __restrict__ Wr,
                  const float* __restrict__ Wd,
                  const float* __restrict__ bd,
                  float* __restrict__ out)
{
    __shared__ _Float16 hs[2][16][HPAD];   // double-buffered h (f16)

    const int tid    = threadIdx.x;
    const int lane   = tid & 31;
    const int w      = tid >> 5;
    const int mbase  = blockIdx.x * 16;
    const int n      = lane & 15;
    const int m      = lane & 15;          // A-frag M
    const int hk     = (lane >> 4) * 8;
    const int rowoff = (lane < 16) ? 0 : 8;
    const int u      = w * 16 + n;         // this lane's hidden unit

    // Wr fragments in registers: brf[kt][g], gate g cols = g*128 + w*16 + i.
    v16h brf[4][4];
    #pragma unroll
    for (int kt = 0; kt < 4; ++kt) {
        #pragma unroll
        for (int g = 0; g < 4; ++g) {
            int n0 = g * UNITS + w * 16;
            v16h f;
            #pragma unroll
            for (int i = 0; i < 16; ++i) {
                int k = kt * 32 + ((i < 8) ? (hk + i) : (16 + hk + (i - 8)));
                f[i] = (_Float16)Wr[k * GATES + n0 + n];
            }
            brf[kt][g] = f;
        }
    }

    // h0 = 0 (buffer 0)
    for (int i = tid; i < 16 * UNITS; i += 256)
        hs[0][i >> 7][i & 127] = (_Float16)0.f;

    float creg[8];
    #pragma unroll
    for (int r = 0; r < 8; ++r) creg[r] = 0.f;

    // Pipeline: preload xg for t=0.
    v8f xn[4];
    #pragma unroll
    for (int g = 0; g < 4; ++g)
        #pragma unroll
        for (int r = 0; r < 8; ++r)
            xn[g][r] = (float)xg[((long)0 * BATCH + mbase + r + rowoff) * GATES
                                 + g * UNITS + u];

    for (int t = 0; t < TLEN; ++t) {
        __syncthreads();   // h(t) visible to every wave
        const _Float16 (*hin)[HPAD] = hs[t & 1];

        v8f acc[4];
        #pragma unroll
        for (int g = 0; g < 4; ++g) acc[g] = xn[g];

        // z = xg_t + h @ Wr : 16 WMMAs, 4 independent chains (ILP=4).
        #pragma unroll
        for (int kt = 0; kt < 4; ++kt) {
            const _Float16* ap = &hin[m][kt * 32 + hk];
            v16h afrag = load_frag16(ap, ap + 16);
            #pragma unroll
            for (int g = 0; g < 4; ++g)
                acc[g] = __builtin_amdgcn_wmma_f32_16x16x32_f16(
                    false, afrag, false, brf[kt][g], (short)0, acc[g],
                    false, false);
        }

        // Prefetch next step's xg while gates run.
        if (t + 1 < TLEN) {
            #pragma unroll
            for (int g = 0; g < 4; ++g)
                #pragma unroll
                for (int r = 0; r < 8; ++r)
                    xn[g][r] = (float)xg[((long)(t + 1) * BATCH
                                          + mbase + r + rowoff) * GATES
                                         + g * UNITS + u];
        }

        // Gates (all register-local): i,f,o sigmoid; g relu.
        _Float16 (*hout)[HPAD] = hs[(t + 1) & 1];
        #pragma unroll
        for (int r = 0; r < 8; ++r) {
            float ig = fast_sigmoid(acc[0][r]);
            float fg = fast_sigmoid(acc[1][r]);
            float gg = acc[2][r] > 0.f ? acc[2][r] : 0.f;
            float og = fast_sigmoid(acc[3][r]);
            float c  = fg * creg[r] + ig * gg;
            creg[r]  = c;
            hout[r + rowoff][u] = (_Float16)(og * (c > 0.f ? c : 0.f));
        }
    }
    __syncthreads();

    // Dense head: out[16,2] = h_last @ Wd + bd. TLEN even -> h_last in hs[0].
    if (tid < 32) {
        int mm = tid >> 1, oo = tid & 1;
        float s = bd[oo];
        for (int uu = 0; uu < UNITS; ++uu)
            s += (float)hs[0][mm][uu] * Wd[uu * 2 + oo];
        out[(mbase + mm) * 2 + oo] = s;
    }
}

// ---------------------------------------------------------------------------
extern "C" void kernel_launch(void* const* d_in, const int* in_sizes, int n_in,
                              void* d_out, int out_size, void* d_ws, size_t ws_size,
                              hipStream_t stream) {
    (void)in_sizes; (void)n_in; (void)out_size; (void)ws_size;
    const int*   x    = (const int*)d_in[0];
    const float* emb  = (const float*)d_in[1];
    const float* Wk   = (const float*)d_in[2];
    const float* Wr   = (const float*)d_in[3];
    const float* bias = (const float*)d_in[4];
    const float* Wd   = (const float*)d_in[5];
    const float* bd   = (const float*)d_in[6];
    float*       out  = (float*)d_out;

    // Workspace: xg [T][B][512] f16 (256MB) | Wkh f16 (256KB) | embh f16 (5MB).
    char* ws = (char*)d_ws;
    _Float16* xg   = (_Float16*)ws;
    _Float16* Wkh  = (_Float16*)(ws + (size_t)TLEN * BATCH * GATES * 2);
    _Float16* embh = (_Float16*)(ws + (size_t)TLEN * BATCH * GATES * 2
                                    + (size_t)EMB * GATES * 2);

    f32_to_f16_kernel<<<dim3((EMB * GATES + 255) / 256), dim3(256), 0, stream>>>(
        Wk, Wkh, EMB * GATES);
    f32_to_f16_kernel<<<dim3((VOCAB * EMB + 255) / 256), dim3(256), 0, stream>>>(
        emb, embh, VOCAB * EMB);

    size_t shmem1 = (size_t)(64 * EMB + 32 * GATES) * sizeof(_Float16);
    lstm_xgemm_kernel<<<dim3((BATCH * TLEN) / 64), dim3(256), shmem1, stream>>>(
        x, embh, Wkh, bias, xg);

    lstm_recur_kernel<<<dim3(BATCH / 16), dim3(256), 0, stream>>>(
        xg, Wr, Wd, bd, out);
}
